// BinaryBlock_15307263443110
// MI455X (gfx1250) — 8-bit matrix operand order repaired and checked by reading, compiled, not run on hardware
//
#include <hip/hip_runtime.h>

typedef int v8i __attribute__((ext_vector_type(8)));

#define BSZ   16
#define CIN   64
#define COUTC 64
#define KW    3
#define LEN   65536
#define STRIP 128
#define XSTRIDE 72
#define STRIPS_PER_WG 8
#define NSTRIPS_PER_B (LEN / STRIP)

__device__ __forceinline__ int kbaseA(int v, int hi) {
  return ((v & 1) * 4) + (((v >> 1) & 1) * 16) + ((v >> 2) * 32) + hi * 8;
}
__device__ __forceinline__ int kbaseB(int v, int hi) {
  return ((v & 1) * 4) + (((v >> 1) & 1) * 16) + ((v >> 2) * 32) + hi * 8;
}

__global__ __launch_bounds__(256)
void binconv1d_iu8_wmma(const float* __restrict__ x, const float* __restrict__ w,
                        const float* __restrict__ bias, const float* __restrict__ alpha,
                        const float* __restrict__ beta, const float* __restrict__ gam,
                        const float* __restrict__ zeta, float* __restrict__ out) {
  __shared__ signed char lds_w[KW * COUTC * CIN];
  __shared__ signed char lds_x[(STRIP + 2) * XSTRIDE];
  __shared__ float lds_bias[COUTC], lds_beta[COUTC], lds_gamma[COUTC],
                   lds_zeta[COUTC], lds_alpha[CIN];

  const int tid  = threadIdx.x;
  const int lane = tid & 31;
  const int wave = tid >> 5;
  const int n    = lane & 15;
  const int hi   = lane >> 4;

  if (tid < COUTC) {
    lds_bias[tid]  = bias[tid];
    lds_beta[tid]  = beta[tid];
    lds_gamma[tid] = gam[tid];
    lds_zeta[tid]  = zeta[tid];
    lds_alpha[tid] = alpha[tid];
  }
  for (int idx = tid; idx < KW * COUTC * CIN; idx += 256) {
    int cout = idx / (CIN * KW);
    int rem  = idx - cout * (CIN * KW);
    int cin  = rem / KW;
    int k    = rem - cin * KW;
    lds_w[(k * COUTC + cout) * CIN + cin] = (w[idx] >= 0.0f) ? 1 : -1;
  }
  __syncthreads();

  v8i a[KW][4];
#pragma unroll
  for (int k = 0; k < KW; ++k)
#pragma unroll
    for (int t = 0; t < 4; ++t)
#pragma unroll
      for (int v = 0; v < 8; ++v)
        a[k][t][v] = *(const int*)&lds_w[(k * COUTC + t * 16 + n) * CIN + kbaseA(v, hi)];

  float pbias[4][8], pbeta[4][8], pgam[4][8], pzeta[4][8];
#pragma unroll
  for (int t = 0; t < 4; ++t)
#pragma unroll
    for (int r = 0; r < 8; ++r) {
      int cout = t * 16 + hi * 8 + r;
      pbias[t][r] = lds_bias[cout];
      pbeta[t][r] = lds_beta[cout];
      pgam[t][r]  = lds_gamma[cout];
      pzeta[t][r] = lds_zeta[cout];
    }

  const int strip0 = blockIdx.x * STRIPS_PER_WG;
  const int b      = strip0 / NSTRIPS_PER_B;
  const size_t xbase_b = (size_t)b * CIN * LEN;

  for (int it = 0; it < STRIPS_PER_WG; ++it) {
    const int l0 = ((strip0 + it) & (NSTRIPS_PER_B - 1)) * STRIP;

#pragma unroll 4
    for (int j = 0; j < (CIN * STRIP) / 256; ++j) {
      int idx = j * 256 + tid;
      int cin = idx >> 7;
      int pos = idx & (STRIP - 1);
      float xv = x[xbase_b + (size_t)cin * LEN + l0 + pos];
      lds_x[(pos + 1) * XSTRIDE + cin] = (xv >= lds_alpha[cin]) ? 1 : -1;
    }
    if (tid < 2 * CIN) {
      int cin  = tid >> 1;
      int side = tid & 1;
      int gpos = side ? (l0 + STRIP) : (l0 - 1);
      int p    = side ? (STRIP + 1) : 0;
      signed char v = 0;
      if (gpos >= 0 && gpos < LEN)
        v = (x[xbase_b + (size_t)cin * LEN + gpos] >= lds_alpha[cin]) ? 1 : -1;
      lds_x[p * XSTRIDE + cin] = v;
    }
    if (it + 1 < STRIPS_PER_WG) {
      const int l0n = l0 + STRIP;
      __builtin_prefetch(&x[xbase_b + (size_t)(tid & 63) * LEN + l0n + (tid >> 6) * 32], 0, 0);
    }
    __syncthreads();

    v8i acc[4] = {};
#pragma unroll
    for (int k = 0; k < KW; ++k) {
      v8i bm;
      const int p = wave * 16 + n + k;
#pragma unroll
      for (int v = 0; v < 8; ++v)
        bm[v] = *(const int*)&lds_x[p * XSTRIDE + kbaseB(v, hi)];
#pragma unroll
      for (int t = 0; t < 4; ++t)
        acc[t] = __builtin_amdgcn_wmma_i32_16x16x64_iu8(
             true, a[k][t],  true, bm, acc[t],
             false,  false);
    }

    const int opos = l0 + wave * 16 + n;
#pragma unroll
    for (int t = 0; t < 4; ++t) {
#pragma unroll
      for (int r = 0; r < 8; ++r) {
        int cout = t * 16 + hi * 8 + r;
        float y  = (float)acc[t][r] + pbias[t][r];
        float res = fmaf(pbeta[t][r], fmaxf(y, 0.0f),
                         fmaf(pgam[t][r], fminf(y, 0.0f), pzeta[t][r]));
        out[((size_t)b * COUTC + cout) * LEN + opos] = res;
      }
    }
    __syncthreads();
  }
}

extern "C" void kernel_launch(void* const* d_in, const int* in_sizes, int n_in,
                              void* d_out, int out_size, void* d_ws, size_t ws_size,
                              hipStream_t stream) {
  (void)in_sizes; (void)n_in; (void)out_size; (void)d_ws; (void)ws_size;
  const float* x     = (const float*)d_in[0];
  const float* w     = (const float*)d_in[1];
  const float* bias  = (const float*)d_in[2];
  const float* alpha = (const float*)d_in[3];
  const float* beta  = (const float*)d_in[4];
  const float* gam   = (const float*)d_in[5];
  const float* zeta  = (const float*)d_in[6];
  float* out = (float*)d_out;

  const int total_strips = BSZ * (LEN / STRIP);
  dim3 grid(total_strips / STRIPS_PER_WG);
  dim3 block(256);
  binconv1d_iu8_wmma<<<grid, block, 0, stream>>>(x, w, bias, alpha, beta, gam, zeta, out);
}
